// PropNetNom_35862976922293
// MI455X (gfx1250) — matrix operand order repaired and checked by reading, not compiled, not run on hardware
//
#include <hip/hip_runtime.h>
#include <hip/hip_bf16.h>

// LSTM: B=128, T=2048, H=256, IN=1. PyTorch gate order i,f,g,o.
//   * 8 workgroups x 512 threads (16 wave32). Each WG owns a 16-batch tile and
//     runs the whole T=2048 recurrence internally (no cross-WG sync).
//   * Per step: [16x256] @ [256x1024] as v_wmma_f32_16x16x32_bf16
//     (4 N-tiles x 8 K-tiles per wave = 32 WMMA / wave / step).
//   * W_hh pre-packed to bf16 in exact WMMA-B lane layout with gate rows
//     interleaved (col m -> unit m/4, gate m%4): 512 KB, L2-resident,
//     streamed as coalesced global_load_b128 every step.  A 32-bit zero
//     offset is laundered through inline-asm each timestep so LICM cannot
//     hoist the loads (keeps pointer provenance -> GLOBAL, not FLAT, so the
//     B stream does not touch DScnt).
//   * B fragments triple-buffered (prefetch kt+2 while kt's WMMAs issue),
//     A fragments double-buffered from LDS; sched_barrier(0) fences keep the
//     prefetch groups ahead of the WMMA groups.
//   * h (bf16) double-buffered in LDS; gates spill to wave-private LDS
//     scratch for the lane transpose; exactly one __syncthreads per step.
//   * FC head: wave w reduces batch-row w with __shfl_xor, sigmoid, store.

#define B_    128
#define T_    2048
#define H_    256
#define G4_   1024              // 4*H
#define WAVES 16
#define HROW  264               // h row stride in bf16 elems (256 + 8 pad)
#define GROW  68                // gate scratch row stride in f32 (64 + 4 pad)

typedef __bf16 bf16_t;
typedef __bf16 bf16x16 __attribute__((ext_vector_type(16)));
typedef float  f32x8   __attribute__((ext_vector_type(8)));
typedef unsigned int uint32x4 __attribute__((ext_vector_type(4)));

union Frag16 { bf16x16 v; uint32x4 q[2]; };
union H8     { uint32x4 q; bf16_t h[8]; };

__device__ __forceinline__ void sched_fence() {
#if __has_builtin(__builtin_amdgcn_sched_barrier)
    __builtin_amdgcn_sched_barrier(0);
#endif
}

__device__ __forceinline__ float fast_rcp(float v) {
#if __has_builtin(__builtin_amdgcn_rcpf)
    return __builtin_amdgcn_rcpf(v);
#else
    return 1.0f / v;
#endif
}
__device__ __forceinline__ float fast_sig(float v) {
    return fast_rcp(1.0f + __expf(-v));
}
__device__ __forceinline__ float fast_tanh(float v) {
#if __has_builtin(__builtin_amdgcn_tanhf)
    return __builtin_amdgcn_tanhf(v);
#else
    float e = __expf(-2.0f * v);
    return (1.0f - e) * fast_rcp(1.0f + e);
#endif
}

// Load the 4 B fragments (one per N-tile) of K-tile kt. bt already includes
// this lane's +L*16 offset; fragment (nt,kt) starts at element (nt*8+kt)*512.
__device__ __forceinline__ void load_bfrag(Frag16 (&dst)[4],
                                           const bf16_t* bt, int kt) {
    #pragma unroll
    for (int nt = 0; nt < 4; ++nt) {
        const uint32x4* bp = (const uint32x4*)(bt + (nt * 8 + kt) * 512);
        dst[nt].q[0] = bp[0];
        dst[nt].q[1] = bp[1];
    }
}
// Load the A fragment of K-tile kt from the LDS h row (ap0 already includes
// +Lm*HROW + 8*Lh; lane chunks are [32kt .. +7] and [32kt+16 .. +7]).
__device__ __forceinline__ void load_afrag(Frag16& A,
                                           const bf16_t* ap0, int kt) {
    const bf16_t* ap = ap0 + 32 * kt;
    A.q[0] = *(const uint32x4*)(ap);
    A.q[1] = *(const uint32x4*)(ap + 16);
}

// ---------------------------------------------------------------------------
// Prep: pack W_hh -> bf16 WMMA-B fragments; permute W_ih / (b_ih+b_hh).
// Bpack element index e = ((ct*8 + kt)*32 + L)*16 + j
//   holds W_hh[row][k], m = 16*ct + (L%16), row = (m%4)*256 + m/4,
//   k = 32*kt + 8*(L/16) + (j%8) + 16*(j/8)   (table order, matches A).
// ---------------------------------------------------------------------------
__global__ void prep_pack(const float* __restrict__ W_ih,
                          const float* __restrict__ W_hh,
                          const float* __restrict__ b_ih,
                          const float* __restrict__ b_hh,
                          bf16_t* __restrict__ Bpack,
                          float*  __restrict__ xw_perm,
                          float*  __restrict__ bias_perm)
{
    int e = blockIdx.x * blockDim.x + threadIdx.x;   // 0 .. 262143
    int j  = e & 15;
    int L  = (e >> 4) & 31;
    int kt = (e >> 9) & 7;
    int ct = e >> 12;
    int m  = 16 * ct + (L & 15);
    int row = (m & 3) * H_ + (m >> 2);
    int k   = 32 * kt + 8 * (L >> 4) + (j & 7) + 16 * (j >> 3);
    Bpack[e] = (bf16_t)W_hh[row * H_ + k];
    if (e < G4_) {
        int r2 = (e & 3) * H_ + (e >> 2);
        xw_perm[e]   = W_ih[r2];                 // IN == 1
        bias_perm[e] = b_ih[r2] + b_hh[r2];
    }
}

// ---------------------------------------------------------------------------
// Main recurrent kernel: one 16-batch tile per workgroup.
// ---------------------------------------------------------------------------
__global__ __launch_bounds__(512)
void lstm_main(const float*  __restrict__ x,        // [128][2048]
               const bf16_t* __restrict__ Bpack,
               const float*  __restrict__ xw_perm,
               const float*  __restrict__ bias_perm,
               const float*  __restrict__ W_fc,     // [256]
               const float*  __restrict__ b_fc,     // [1]
               float*        __restrict__ out)      // [128][2048]
{
    __shared__ bf16_t hbuf[2 * 16 * HROW];          // double-buffered h, 16.5 KB
    __shared__ float  gsc[WAVES * 16 * GROW];       // wave-private gate scratch, 68 KB

    const int tid = threadIdx.x;
    const int w   = tid >> 5;          // wave id: owns units [16w,16w+16)
    const int L   = tid & 31;
    const int Lm  = L & 15;
    const int Lh  = L >> 4;            // 0/1
    const int bg0 = blockIdx.x * 16;   // first global batch of this tile

    // h0 = 0
    for (int i = tid; i < 2 * 16 * HROW; i += 512) hbuf[i] = (bf16_t)0.0f;

    // per-wave constants: x-weight & bias per owned column (m = 64w+16nt+Lm)
    float xw[4], bs[4];
    #pragma unroll
    for (int nt = 0; nt < 4; ++nt) {
        int m = 64 * w + 16 * nt + Lm;
        xw[nt] = xw_perm[m];
        bs[nt] = bias_perm[m];
    }
    // FC weights: lane L holds W_fc[8L .. 8L+7]
    float wfc[8];
    #pragma unroll
    for (int j = 0; j < 8; ++j) wfc[j] = W_fc[8 * L + j];
    const float bfc = b_fc[0];

    float cst[8];                       // cell state: 8 (unit,batch) pairs/lane
    #pragma unroll
    for (int q = 0; q < 8; ++q) cst[q] = 0.0f;

    float* gw = gsc + w * 16 * GROW;    // this wave's gate scratch

    // Wave's slice of packed B: col-tiles ct = 4w .. 4w+3.
    const bf16_t* bwave = Bpack + (size_t)w * 16384 + (size_t)L * 16;

    __syncthreads();

    int p = 0;
    for (int t = 0; t < T_; ++t) {
        // Launder a zero 32-bit offset: loads stay inside the loop (no LICM
        // hoist / scratch spill) while keeping GLOBAL address-space inference.
        unsigned bofs = 0;
        asm volatile("" : "+v"(bofs));
        const bf16_t* bt = bwave + bofs;

        // ---- exact x contribution folded into C init ----
        float xv[8];
        #pragma unroll
        for (int v = 0; v < 8; ++v)
            xv[v] = x[(bg0 + v + 8 * Lh) * T_ + t];   // row M = v + 8*Lh

        f32x8 C[4];
        #pragma unroll
        for (int nt = 0; nt < 4; ++nt)
            #pragma unroll
            for (int v = 0; v < 8; ++v)
                C[nt][v] = xv[v] * xw[nt] + bs[nt];

        // ---- gates += h @ W_hh^T : depth-2 pipelined B, depth-1 A ----
        const bf16_t* hp  = hbuf + p * 16 * HROW;
        const bf16_t* ap0 = hp + Lm * HROW + 8 * Lh;

        Frag16 B0[4], B1[4], B2[4];
        Frag16 Aa, Ab;
        load_bfrag(B0, bt, 0);
        load_bfrag(B1, bt, 1);
        load_afrag(Aa, ap0, 0);

        #pragma unroll
        for (int kt = 0; kt < 8; ++kt) {
            Frag16 (&cur)[4] = (kt % 3 == 0) ? B0 : (kt % 3 == 1) ? B1 : B2;
            if (kt < 6) {
                Frag16 (&nx2)[4] =
                    ((kt + 2) % 3 == 0) ? B0 : ((kt + 2) % 3 == 1) ? B1 : B2;
                load_bfrag(nx2, bt, kt + 2);
            }
            if (kt < 7) {
                Frag16& An = (kt & 1) ? Aa : Ab;
                load_afrag(An, ap0, kt + 1);
            }
            sched_fence();          // keep prefetch group ahead of WMMA group
            Frag16& Ac = (kt & 1) ? Ab : Aa;
            #pragma unroll
            for (int nt = 0; nt < 4; ++nt)
                C[nt] = __builtin_amdgcn_wmma_f32_16x16x32_bf16(
                    false, Ac.v, false, cur[nt].v, (short)0, C[nt], false, false);
            sched_fence();
        }

        // ---- spill gates to wave-private LDS (lane transpose) ----
        #pragma unroll
        for (int nt = 0; nt < 4; ++nt)
            #pragma unroll
            for (int v = 0; v < 8; ++v)
                gw[(v + 8 * Lh) * GROW + nt * 16 + Lm] = C[nt][v];

        // ---- elementwise LSTM update: lane owns (batch Lm, units 8*Lh+q) ----
        bf16_t* hn = hbuf + (p ^ 1) * 16 * HROW;
        const int b = Lm;
        #pragma unroll
        for (int q = 0; q < 8; ++q) {
            const int ul = 8 * Lh + q;
            const float4 g4 = *(const float4*)(gw + b * GROW + 4 * ul); // i,f,g,o
            float ig = fast_sig(g4.x);
            float fg = fast_sig(g4.y);
            float gg = fast_tanh(g4.z);
            float og = fast_sig(g4.w);
            float c  = fg * cst[q] + ig * gg;
            cst[q] = c;
            float h = og * fast_tanh(c);
            hn[b * HROW + 16 * w + ul] = (bf16_t)h;
        }

        __syncthreads();   // the only barrier per timestep

        // ---- FC head: wave w reduces batch row w of the new h ----
        {
            H8 hv;
            hv.q = *(const uint32x4*)(hn + w * HROW + 8 * L);
            float s = 0.0f;
            #pragma unroll
            for (int j = 0; j < 8; ++j) s += (float)hv.h[j] * wfc[j];
            s += __shfl_xor(s, 16);
            s += __shfl_xor(s, 8);
            s += __shfl_xor(s, 4);
            s += __shfl_xor(s, 2);
            s += __shfl_xor(s, 1);
            if (L == 0)
                out[(bg0 + w) * T_ + t] = fast_sig(s + bfc);
        }
        p ^= 1;
    }
}

// ---------------------------------------------------------------------------
extern "C" void kernel_launch(void* const* d_in, const int* in_sizes, int n_in,
                              void* d_out, int out_size, void* d_ws, size_t ws_size,
                              hipStream_t stream) {
    (void)in_sizes; (void)n_in; (void)out_size; (void)ws_size;
    const float* x    = (const float*)d_in[0];
    const float* W_ih = (const float*)d_in[1];
    const float* W_hh = (const float*)d_in[2];
    const float* b_ih = (const float*)d_in[3];
    const float* b_hh = (const float*)d_in[4];
    const float* W_fc = (const float*)d_in[5];
    const float* b_fc = (const float*)d_in[6];
    float* out = (float*)d_out;

    bf16_t* Bpack     = (bf16_t*)d_ws;                       // 512 KB
    float*  xw_perm   = (float*)((char*)d_ws + 512 * 1024);  // 4 KB
    float*  bias_perm = xw_perm + G4_;                       // 4 KB

    // Pack W_hh into WMMA-B bf16 fragments + permuted x-weights/bias.
    prep_pack<<<1024, 256, 0, stream>>>(W_ih, W_hh, b_ih, b_hh,
                                        Bpack, xw_perm, bias_perm);

    // 8 batch tiles, one workgroup (16 wave32) each.
    lstm_main<<<8, 512, 0, stream>>>(x, Bpack, xw_perm, bias_perm,
                                     W_fc, b_fc, out);
}
